// Llama2_76459007803576
// MI455X (gfx1250) — matrix operand order repaired and checked by reading, not compiled, not run on hardware
//
#include <hip/hip_runtime.h>

typedef unsigned short u16;
typedef __attribute__((ext_vector_type(16))) __bf16 bf16x16;
typedef __attribute__((ext_vector_type(8)))  float  f32x8;

#define SEQ   2048
#define DIM   1024
#define NH    16
#define HD    64
#define NL    2
#define HID   2816
#define VOCAB 32000

union Frag { bf16x16 v; uint4 q[2]; };

__device__ __forceinline__ u16 f2bf(float f) {
    union { float f; unsigned u; } x; x.f = f;
    unsigned r = (x.u + 0x7FFFu + ((x.u >> 16) & 1u)) >> 16;
    return (u16)r;
}
__device__ __forceinline__ float bf2f(u16 b) {
    union { unsigned u; float f; } x; x.u = ((unsigned)b) << 16;
    return x.f;
}

// Per-lane async copy of 16B from global to LDS (ASYNCcnt-tracked DMA path).
// ISA §10.2: flat LDS addresses carry the LDS byte offset in addr[31:0], so the
// low 32 bits of a __shared__ pointer are the VDST operand.
__device__ __forceinline__ void async_g2l_b128(void* lds_dst, const void* gsrc) {
    unsigned ldsoff = (unsigned)(size_t)lds_dst;
    unsigned long long ga = (unsigned long long)(size_t)gsrc;
    asm volatile("global_load_async_to_lds_b128 %0, %1, off"
                 :: "v"(ldsoff), "v"(ga) : "memory");
}

// ---------------------------------------------------------------------------
// Weight convert + transpose: fp32 [K,N] -> bf16 [N,K]
// ---------------------------------------------------------------------------
__global__ __launch_bounds__(256) void convT_kernel(const float* __restrict__ in,
                                                    u16* __restrict__ out,
                                                    int K, int N) {
    int idx = blockIdx.x * 256 + threadIdx.x;
    if (idx >= K * N) return;
    int k = idx / N, n = idx % N;
    out[(size_t)n * K + k] = f2bf(in[idx]);
}

// ---------------------------------------------------------------------------
// Embedding gather: h[t,d] = w_out[d, tok[t]]   (fp32)
// ---------------------------------------------------------------------------
__global__ __launch_bounds__(256) void embed_kernel(const int* __restrict__ tokens,
                                                    const float* __restrict__ w_out,
                                                    float* __restrict__ h) {
    int t = blockIdx.x;
    int d = threadIdx.x * 4;
    int tok = tokens[t];
#pragma unroll
    for (int e = 0; e < 4; ++e)
        h[t * DIM + d + e] = w_out[(size_t)(d + e) * VOCAB + tok];
}

// ---------------------------------------------------------------------------
// RMSNorm fp32 row -> bf16 row, multiplied by weight
// ---------------------------------------------------------------------------
__global__ __launch_bounds__(256) void rmsnorm_kernel(const float* __restrict__ x,
                                                      const float* __restrict__ w,
                                                      u16* __restrict__ out) {
    int row = blockIdx.x, tid = threadIdx.x;
    const float* xr = x + (size_t)row * DIM;
    float4 v = *(const float4*)(xr + tid * 4);
    float ss = v.x * v.x + v.y * v.y + v.z * v.z + v.w * v.w;
#pragma unroll
    for (int m = 16; m >= 1; m >>= 1) ss += __shfl_xor(ss, m, 32);
    __shared__ float part[8];
    if ((tid & 31) == 0) part[tid >> 5] = ss;
    __syncthreads();
    float tot = 0.f;
#pragma unroll
    for (int i = 0; i < 8; ++i) tot += part[i];
    float sc = rsqrtf(tot / (float)DIM + 1e-5f);
    float4 wv = *(const float4*)(w + tid * 4);
    uint2 pk;
    pk.x = (unsigned)f2bf(v.x * sc * wv.x) | ((unsigned)f2bf(v.y * sc * wv.y) << 16);
    pk.y = (unsigned)f2bf(v.z * sc * wv.z) | ((unsigned)f2bf(v.w * sc * wv.w) << 16);
    *(uint2*)(out + (size_t)row * DIM + tid * 4) = pk;
}

// ---------------------------------------------------------------------------
// RoPE on q (with 1/sqrt(HD) folded in) and k, bf16 in-place
// ---------------------------------------------------------------------------
__global__ __launch_bounds__(256) void rope_kernel(u16* __restrict__ q,
                                                   u16* __restrict__ k) {
    int idx = blockIdx.x * 256 + threadIdx.x;
    if (idx >= SEQ * NH * (HD / 2)) return;
    int t   = idx >> 9;          // / (NH*32)
    int rem = idx & 511;
    int h   = rem >> 5;
    int i   = rem & 31;
    float freq = __powf(10000.0f, -(float)(2 * i) / (float)HD);
    float ang  = (float)t * freq;
    float s, c; __sincosf(ang, &s, &c);
    int base = t * DIM + h * HD + 2 * i;
    float xr = bf2f(q[base]), xi = bf2f(q[base + 1]);
    q[base]     = f2bf((xr * c - xi * s) * 0.125f);   // fold softmax scale
    q[base + 1] = f2bf((xr * s + xi * c) * 0.125f);
    xr = bf2f(k[base]); xi = bf2f(k[base + 1]);
    k[base]     = f2bf(xr * c - xi * s);
    k[base + 1] = f2bf(xr * s + xi * c);
}

// ---------------------------------------------------------------------------
// SiLU(g) * u elementwise, bf16
// ---------------------------------------------------------------------------
__global__ __launch_bounds__(256) void silu_mul_kernel(const u16* __restrict__ g,
                                                       const u16* __restrict__ u,
                                                       u16* __restrict__ a, int n) {
    int idx = blockIdx.x * 256 + threadIdx.x;
    if (idx >= n) return;
    float gf = bf2f(g[idx]);
    float s  = gf / (1.f + __expf(-gf));
    a[idx] = f2bf(s * bf2f(u[idx]));
}

// ---------------------------------------------------------------------------
// WMMA GEMM: C[M,N] = A[M,K](bf16,row-major) x Wt[N,K](bf16, transposed weight)
// 256 threads = 8 waves (2x4); wave computes a 64x64 C block.
// MODE 0: store bf16; MODE 1: fp32 +=; MODE 2: fp32 store.
// ---------------------------------------------------------------------------
template <int MODE>
__global__ __launch_bounds__(256) void gemm_bf16_kernel(const u16* __restrict__ A,
                                                        const u16* __restrict__ Wt,
                                                        void* __restrict__ Cv,
                                                        int M, int N, int K) {
    int tid  = threadIdx.x;
    int lane = tid & 31, wid = tid >> 5;
    int l15  = lane & 15, lh = lane >> 4;
    int bm = blockIdx.y * 128 + (wid >> 2) * 64;
    int bn = blockIdx.x * 256 + (wid & 3) * 64;

    f32x8 acc[4][4] = {};
    const u16* pA[4];
    const u16* pB[4];
#pragma unroll
    for (int mt = 0; mt < 4; ++mt) pA[mt] = A  + (size_t)(bm + mt * 16 + l15) * K;
#pragma unroll
    for (int nt = 0; nt < 4; ++nt) pB[nt] = Wt + (size_t)(bn + nt * 16 + l15) * K;
    int aoff = lh * 16;   // A: half-wave picks K 0..15 vs 16..31
    int boff = lh * 16;   // B: half-wave picks K 0..15 vs 16..31

    for (int k0 = 0; k0 < K; k0 += 32) {
        Frag a[4], b[4];
#pragma unroll
        for (int mt = 0; mt < 4; ++mt) {
            const u16* p = pA[mt] + k0 + aoff;
            a[mt].q[0] = *(const uint4*)p;
            a[mt].q[1] = *(const uint4*)(p + 8);
        }
#pragma unroll
        for (int nt = 0; nt < 4; ++nt) {
            const u16* p = pB[nt] + k0 + boff;
            b[nt].q[0] = *(const uint4*)p;
            b[nt].q[1] = *(const uint4*)(p + 8);
            // warm L0 for the weight stream a few K-tiles ahead
            // (speculative global_prefetch_b8; invalid tails are dropped)
            __builtin_prefetch(p + 128, 0, 3);
        }
#pragma unroll
        for (int mt = 0; mt < 4; ++mt)
#pragma unroll
            for (int nt = 0; nt < 4; ++nt)
                acc[mt][nt] = __builtin_amdgcn_wmma_f32_16x16x32_bf16(
                    false, a[mt].v, false, b[nt].v, (short)0, acc[mt][nt], false, false);
    }

#pragma unroll
    for (int mt = 0; mt < 4; ++mt)
#pragma unroll
        for (int nt = 0; nt < 4; ++nt)
#pragma unroll
            for (int v = 0; v < 8; ++v) {
                int r   = bm + mt * 16 + v + 8 * lh;
                int col = bn + nt * 16 + l15;
                size_t idx = (size_t)r * N + col;
                float val = acc[mt][nt][v];
                if (MODE == 0)      ((u16*)Cv)[idx]   = f2bf(val);
                else if (MODE == 1) ((float*)Cv)[idx] += val;
                else                ((float*)Cv)[idx] = val;
            }
}

// ---------------------------------------------------------------------------
// Flash attention (causal, online softmax), WMMA for QK^T and PV.
// Grid: (SEQ/128 query blocks, NH heads). Block: 256 thr = 8 waves,
// each wave owns 16 query rows. Key blocks of 32:
//   - K tile staged by async DMA (global_load_async_to_lds_b128, ASYNCcnt)
//   - V tile transposed through VGPRs (overlaps with the K DMA)
// ---------------------------------------------------------------------------
__global__ __launch_bounds__(256) void flash_attn_kernel(const u16* __restrict__ Q,
                                                         const u16* __restrict__ Kk,
                                                         const u16* __restrict__ Vv,
                                                         u16* __restrict__ O) {
    __shared__ u16 kbuf[32 * 64];       // [key][dim] row-major
    __shared__ u16 vtbuf[64 * 32];      // [dim][key] (V transposed)
    __shared__ u16 pbuf[8 * 16 * 32];   // per-wave P tile [16 q rows][32 keys]

    int qb = blockIdx.x, h = blockIdx.y;
    int tid = threadIdx.x, wid = tid >> 5, lane = tid & 31;
    int l15 = lane & 15, lh = lane >> 4;

    // Q A-fragments for this wave's 16 rows (K-dim = 64 -> two frags)
    int qrow = qb * 128 + wid * 16 + l15;
    const u16* qp = Q + (size_t)qrow * DIM + h * HD;
    Frag qa0, qa1;
    {
        int off = lh * 16;
        qa0.q[0] = *(const uint4*)(qp + off);
        qa0.q[1] = *(const uint4*)(qp + 8 + off);
        qa1.q[0] = *(const uint4*)(qp + 32 + off);
        qa1.q[1] = *(const uint4*)(qp + 40 + off);
    }

    f32x8 o[4] = {};
    float m_i[8], l_i[8];
#pragma unroll
    for (int v = 0; v < 8; ++v) { m_i[v] = -1e30f; l_i[v] = 0.f; }

    int stg_r  = tid >> 3;          // staging row (key) for this thread
    int stg_c8 = (tid & 7) * 8;     // staging dim start

    int jend = qb * 128 + 96;             // last key-block start (causal)
    for (int j = 0; j <= jend; j += 32) {
        __syncthreads();
        // --- K tile: async DMA directly into LDS (no VGPR round-trip) ---
        const u16* gk = Kk + (size_t)(j + stg_r) * DIM + h * HD + stg_c8;
        async_g2l_b128(kbuf + stg_r * 64 + stg_c8, gk);
        // --- V tile: load + transpose through VGPRs, overlapping the DMA ---
        {
            uint4 vv = *(const uint4*)(Vv + (size_t)(j + stg_r) * DIM + h * HD + stg_c8);
            const u16* pv = (const u16*)&vv;
#pragma unroll
            for (int e = 0; e < 8; ++e) vtbuf[(stg_c8 + e) * 32 + stg_r] = pv[e];
        }
        // prefetch next key block's K and V rows
        __builtin_prefetch(Kk + (size_t)(j + 32 + stg_r) * DIM + h * HD + stg_c8, 0, 3);
        __builtin_prefetch(Vv + (size_t)(j + 32 + stg_r) * DIM + h * HD + stg_c8, 0, 3);
        asm volatile("s_wait_asynccnt 0" ::: "memory");
        __syncthreads();

        // S = Q x K^T for 32 keys (two 16-key N tiles)
        Frag kb00, kb01, kb10, kb11;
        {
            const u16* p0 = kbuf + (l15)      * 64 + lh * 16;  // keys j..j+15
            const u16* p1 = kbuf + (16 + l15) * 64 + lh * 16;  // keys j+16..j+31
            kb00.q[0] = *(const uint4*)p0;        kb00.q[1] = *(const uint4*)(p0 + 8);
            kb01.q[0] = *(const uint4*)(p0 + 32); kb01.q[1] = *(const uint4*)(p0 + 40);
            kb10.q[0] = *(const uint4*)p1;        kb10.q[1] = *(const uint4*)(p1 + 8);
            kb11.q[0] = *(const uint4*)(p1 + 32); kb11.q[1] = *(const uint4*)(p1 + 40);
        }
        f32x8 s0 = {}, s1 = {};
        s0 = __builtin_amdgcn_wmma_f32_16x16x32_bf16(false, qa0.v, false, kb00.v, (short)0, s0, false, false);
        s0 = __builtin_amdgcn_wmma_f32_16x16x32_bf16(false, qa1.v, false, kb01.v, (short)0, s0, false, false);
        s1 = __builtin_amdgcn_wmma_f32_16x16x32_bf16(false, qa0.v, false, kb10.v, (short)0, s1, false, false);
        s1 = __builtin_amdgcn_wmma_f32_16x16x32_bf16(false, qa1.v, false, kb11.v, (short)0, s1, false, false);

        // online softmax (row = v + 8*lh, col = lane&15 within each 16-key tile)
        int rowb = qb * 128 + wid * 16 + 8 * lh;
        u16* pw = pbuf + wid * 512;
#pragma unroll
        for (int v = 0; v < 8; ++v) {
            int r = rowb + v;
            float a = s0[v]; if (j + l15 > r)      a = -1e30f;
            float b = s1[v]; if (j + 16 + l15 > r) b = -1e30f;
            float mx = fmaxf(a, b);
#pragma unroll
            for (int msk = 1; msk <= 8; msk <<= 1) mx = fmaxf(mx, __shfl_xor(mx, msk, 32));
            float mn = fmaxf(m_i[v], mx);
            float p0 = __expf(a - mn);
            float p1 = __expf(b - mn);
            float sum = p0 + p1;
#pragma unroll
            for (int msk = 1; msk <= 8; msk <<= 1) sum += __shfl_xor(sum, msk, 32);
            float alpha = __expf(m_i[v] - mn);
            l_i[v] = l_i[v] * alpha + sum;
            m_i[v] = mn;
            o[0][v] *= alpha; o[1][v] *= alpha; o[2][v] *= alpha; o[3][v] *= alpha;
            u16* prow = pw + (v + 8 * lh) * 32;
            prow[l15]      = f2bf(p0);
            prow[16 + l15] = f2bf(p1);
        }
        asm volatile("s_wait_dscnt 0" ::: "memory");

        // P A-fragment (16x32) from wave-local LDS tile
        Frag pf;
        {
            const u16* pp = pbuf + wid * 512 + l15 * 32 + lh * 16;
            pf.q[0] = *(const uint4*)pp;
            pf.q[1] = *(const uint4*)(pp + 8);
        }
        // O += P x V  (4 dim tiles of 16)
#pragma unroll
        for (int nt = 0; nt < 4; ++nt) {
            Frag vb;
            const u16* vp = vtbuf + (nt * 16 + l15) * 32 + lh * 16;
            vb.q[0] = *(const uint4*)vp;
            vb.q[1] = *(const uint4*)(vp + 8);
            o[nt] = __builtin_amdgcn_wmma_f32_16x16x32_bf16(
                false, pf.v, false, vb.v, (short)0, o[nt], false, false);
        }
    }

    // normalize and write attention output (bf16)
#pragma unroll
    for (int nt = 0; nt < 4; ++nt)
#pragma unroll
        for (int v = 0; v < 8; ++v) {
            int r = qb * 128 + wid * 16 + v + 8 * lh;
            O[(size_t)r * DIM + h * HD + nt * 16 + l15] = f2bf(o[nt][v] / l_i[v]);
        }
}

// ---------------------------------------------------------------------------
// Host orchestration
// ---------------------------------------------------------------------------
extern "C" void kernel_launch(void* const* d_in, const int* in_sizes, int n_in,
                              void* d_out, int out_size, void* d_ws, size_t ws_size,
                              hipStream_t stream) {
    const int*   tokens = (const int*)d_in[0];
    const float* wq     = (const float*)d_in[1];
    const float* wk     = (const float*)d_in[2];
    const float* wv     = (const float*)d_in[3];
    const float* wo     = (const float*)d_in[4];
    const float* w1     = (const float*)d_in[5];
    const float* w2     = (const float*)d_in[6];
    const float* w3     = (const float*)d_in[7];
    const float* anw    = (const float*)d_in[8];
    const float* fnw    = (const float*)d_in[9];
    const float* finw   = (const float*)d_in[10];
    const float* w_out  = (const float*)d_in[11];

    char* ws = (char*)d_ws;
    auto alloc = [&](size_t bytes) -> void* {
        void* p = (void*)ws;
        ws += (bytes + 255) & ~(size_t)255;
        return p;
    };
    u16* wqT   = (u16*)alloc((size_t)NL * DIM * DIM * 2);
    u16* wkT   = (u16*)alloc((size_t)NL * DIM * DIM * 2);
    u16* wvT   = (u16*)alloc((size_t)NL * DIM * DIM * 2);
    u16* woT   = (u16*)alloc((size_t)NL * DIM * DIM * 2);
    u16* w1T   = (u16*)alloc((size_t)NL * DIM * HID * 2);
    u16* w3T   = (u16*)alloc((size_t)NL * DIM * HID * 2);
    u16* w2T   = (u16*)alloc((size_t)NL * HID * DIM * 2);
    u16* woutT = (u16*)alloc((size_t)DIM * VOCAB * 2);
    float* h   = (float*)alloc((size_t)SEQ * DIM * 4);
    u16* xn    = (u16*)alloc((size_t)SEQ * DIM * 2);
    u16* qb    = (u16*)alloc((size_t)SEQ * DIM * 2);
    u16* kb    = (u16*)alloc((size_t)SEQ * DIM * 2);
    u16* vb    = (u16*)alloc((size_t)SEQ * DIM * 2);
    u16* xat   = (u16*)alloc((size_t)SEQ * DIM * 2);
    u16* gbuf  = (u16*)alloc((size_t)SEQ * HID * 2);
    u16* ubuf  = (u16*)alloc((size_t)SEQ * HID * 2);
    u16* abuf  = (u16*)alloc((size_t)SEQ * HID * 2);

    // --- weight convert + transpose (bf16 [N,K]) ---
    const int nSq = (DIM * DIM + 255) / 256;
    const int nDH = (DIM * HID + 255) / 256;
    for (int l = 0; l < NL; ++l) {
        size_t oS = (size_t)l * DIM * DIM, oH = (size_t)l * DIM * HID;
        convT_kernel<<<nSq, 256, 0, stream>>>(wq + oS, wqT + oS, DIM, DIM);
        convT_kernel<<<nSq, 256, 0, stream>>>(wk + oS, wkT + oS, DIM, DIM);
        convT_kernel<<<nSq, 256, 0, stream>>>(wv + oS, wvT + oS, DIM, DIM);
        convT_kernel<<<nSq, 256, 0, stream>>>(wo + oS, woT + oS, DIM, DIM);
        convT_kernel<<<nDH, 256, 0, stream>>>(w1 + oH, w1T + oH, DIM, HID);
        convT_kernel<<<nDH, 256, 0, stream>>>(w3 + oH, w3T + oH, DIM, HID);
        convT_kernel<<<nDH, 256, 0, stream>>>(w2 + oH, w2T + oH, HID, DIM);
    }
    convT_kernel<<<(DIM * VOCAB + 255) / 256, 256, 0, stream>>>(w_out, woutT, DIM, VOCAB);

    // --- embedding ---
    embed_kernel<<<SEQ, 256, 0, stream>>>(tokens, w_out, h);

    dim3 gDim(DIM / 256, SEQ / 128);    // N=1024
    dim3 gHid(HID / 256, SEQ / 128);    // N=2816
    dim3 gVoc(VOCAB / 256, SEQ / 128);  // N=32000
    dim3 gAtt(SEQ / 128, NH);

    for (int l = 0; l < NL; ++l) {
        size_t oS = (size_t)l * DIM * DIM, oH = (size_t)l * DIM * HID;
        // attention block
        rmsnorm_kernel<<<SEQ, 256, 0, stream>>>(h, anw + l * DIM, xn);
        gemm_bf16_kernel<0><<<gDim, 256, 0, stream>>>(xn, wqT + oS, qb, SEQ, DIM, DIM);
        gemm_bf16_kernel<0><<<gDim, 256, 0, stream>>>(xn, wkT + oS, kb, SEQ, DIM, DIM);
        gemm_bf16_kernel<0><<<gDim, 256, 0, stream>>>(xn, wvT + oS, vb, SEQ, DIM, DIM);
        rope_kernel<<<(SEQ * NH * (HD / 2)) / 256, 256, 0, stream>>>(qb, kb);
        flash_attn_kernel<<<gAtt, 256, 0, stream>>>(qb, kb, vb, xat);
        gemm_bf16_kernel<1><<<gDim, 256, 0, stream>>>(xat, woT + oS, h, SEQ, DIM, DIM);
        // FFN block
        rmsnorm_kernel<<<SEQ, 256, 0, stream>>>(h, fnw + l * DIM, xn);
        gemm_bf16_kernel<0><<<gHid, 256, 0, stream>>>(xn, w1T + oH, gbuf, SEQ, HID, DIM);
        gemm_bf16_kernel<0><<<gHid, 256, 0, stream>>>(xn, w3T + oH, ubuf, SEQ, HID, DIM);
        silu_mul_kernel<<<(SEQ * HID + 255) / 256, 256, 0, stream>>>(gbuf, ubuf, abuf, SEQ * HID);
        gemm_bf16_kernel<1><<<gDim, 256, 0, stream>>>(abuf, w2T + oH, h, SEQ, DIM, HID);
    }

    // final norm + logits
    rmsnorm_kernel<<<SEQ, 256, 0, stream>>>(h, finw, xn);
    gemm_bf16_kernel<2><<<gVoc, 256, 0, stream>>>(xn, woutT, (float*)d_out, SEQ, VOCAB, DIM);
}
